// QuantumQKGenerator6Qubit_47777216201063
// MI455X (gfx1250) — compile-verified, matrix operand order checked by reading, not run on hardware
//
#include <hip/hip_runtime.h>
#include <hip/hip_bf16.h>
#include <math.h>

// MI455X / gfx1250 implementation.
//
// Strategy: fold QFT * (tensor-RY layer) * (CNOT-ring permutation) into a single
// shared 64x64 complex matrix M (computed once per call in a tiny setup kernel),
// so the whole circuit becomes:
//   angles -> per-qubit 2-vectors -> product state psi (64 complex)
//   out_state = M * psi        <-- batched complex GEMM done with v_wmma_f32_16x16x32_f16
//   expvals X/Y/Z per qubit    <-- per-row LDS reduction
//
// The setup kernel writes M as two 128x64 f16 matrices ([Mr;-Mi] and [Mi;Mr])
// pre-swizzled into WMMA B-fragment order in d_ws, so the main kernel loads each
// B fragment with one coalesced 32-byte read per lane.

typedef __attribute__((ext_vector_type(16))) _Float16 v16h;
typedef __attribute__((ext_vector_type(2)))  _Float16 v2h;
typedef __attribute__((ext_vector_type(8)))  float    v8f;

#define NQ  6
#define DIM 64
#define PI_F 3.14159265358979f

// CNOT-ring basis permutation: psi_after[phi(i)] = psi_before[i].
// Bit q of flat index i is (i >> (5-q)) & 1 (wire 0 = MSB, as in the reference).
__device__ __forceinline__ int phi_perm(int i) {
  int b0=(i>>5)&1, b1=(i>>4)&1, b2=(i>>3)&1, b3=(i>>2)&1, b4=(i>>1)&1, b5=i&1;
  int c1=b0^b1, c2=c1^b2, c3=c2^b3, c4=c3^b4, c5=c4^b5;
  int n0=b0^c5;  // = b1^b2^b3^b4^b5
  return (n0<<5)|(c1<<4)|(c2<<3)|(c3<<2)|(c4<<1)|c5;
}

// K index inside a 16-bit 32xK WMMA A/B fragment, per CDNA5 ISA layout:
// lanes 0-15 hold K {0..7,16..23}, lanes 16-31 hold K {8..15,24..31},
// 2 packed halves per VGPR.
__device__ __forceinline__ int frag_k(int ks, int lane, int e) {
  int hi = (lane >> 4) & 1;
  int v = e >> 1, s = e & 1;
  return ks*32 + ((v & 4) ? 16 : 0) + hi*8 + ((v & 3) << 1) + s;
}

// ---------------------------------------------------------------------------
// Setup kernel: build M = F * K * P and emit WMMA-B fragments + trig table.
// ws layout: [0,32768): f16 B-fragments (2 comps x 4 ntiles x 4 ksteps x 32 lanes x 16)
//            [32768, 32768+144): float trig table {cx,sx,cy,sy,cz,sz} per qubit
// ---------------------------------------------------------------------------
__global__ void __launch_bounds__(64)
qqk_setup_kernel(const float* __restrict__ pqcW, void* __restrict__ ws) {
  __shared__ float2 sG[DIM][DIM];  // G = F*K, 32 KB
  const int tid = threadIdx.x;     // 64 threads

  _Float16* wsB   = (_Float16*)ws;
  float*    wsTrig = (float*)((char*)ws + 32768);

  if (tid < NQ) {
    float w0 = pqcW[tid*4+0]*0.5f, w1 = pqcW[tid*4+1]*0.5f, w2 = pqcW[tid*4+2]*0.5f;
    wsTrig[tid*6+0] = cosf(w0); wsTrig[tid*6+1] = sinf(w0);
    wsTrig[tid*6+2] = cosf(w1); wsTrig[tid*6+3] = sinf(w1);
    wsTrig[tid*6+4] = cosf(w2); wsTrig[tid*6+5] = sinf(w2);
  }

  float c3[NQ], s3[NQ];
  #pragma unroll
  for (int q = 0; q < NQ; ++q) {
    float w = pqcW[q*4+3]*0.5f;
    c3[q] = cosf(w); s3[q] = sinf(w);
  }

  // G[n][a] = sum_j F[n][j] * K[j][a],  F[n][j] = exp(+2*pi*i*n*j/64)/8,
  // K[j][a] = prod_q RY(w3_q)[j_q][a_q]  with RY = [[c,-s],[s,c]].
  const int n = tid;  // block is exactly 64 threads
  for (int a = 0; a < DIM; ++a) {
    float gr = 0.f, gi = 0.f;
    for (int j = 0; j < DIM; ++j) {
      float kv = 1.f;
      #pragma unroll
      for (int q = 0; q < NQ; ++q) {
        int bj = (j >> (5-q)) & 1, ba = (a >> (5-q)) & 1;
        float r = (bj == ba) ? c3[q] : (bj ? s3[q] : -s3[q]);
        kv *= r;
      }
      float ang = (float)((n * j) & 63) * (PI_F / 32.0f);
      float fs, fc;
      __sincosf(ang, &fs, &fc);
      gr += kv * fc * 0.125f;
      gi += kv * fs * 0.125f;
    }
    sG[n][a] = make_float2(gr, gi);
  }
  __syncthreads();

  // Emit B fragments. M[n][i] = G[n][phi(i)].
  // out_r = sum_k A[k]*Br[k][n],  A = [psi_r(0..63) | psi_i(0..63)]
  //   Br[k<64][n] =  Re(M[n][k]);   Br[k>=64][n] = -Im(M[n][k-64])
  //   Bi[k<64][n] =  Im(M[n][k]);   Bi[k>=64][n] =  Re(M[n][k-64])
  for (int e_lin = tid; e_lin < 16384; e_lin += 64) {
    int e      = e_lin & 15;
    int lane   = (e_lin >> 4) & 31;
    int fragId = e_lin >> 9;          // (comp*4 + nt)*4 + ks
    int ks   = fragId & 3;
    int nt   = (fragId >> 2) & 3;
    int comp = fragId >> 4;
    int nIdx = nt*16 + (lane & 15);
    int k    = frag_k(ks, lane, e);
    float2 g = sG[nIdx][phi_perm(k & 63)];
    float val;
    if (comp == 0) val = (k < 64) ? g.x : -g.y;
    else           val = (k < 64) ? g.y :  g.x;
    wsB[e_lin] = (_Float16)val;
  }
}

// ---------------------------------------------------------------------------
// Main kernel: 64 threads = 2 waves; each wave handles 32 batch rows (1/lane).
// ---------------------------------------------------------------------------
__global__ void __launch_bounds__(64)
qqk_main_kernel(const float* __restrict__ qae,
                const float* __restrict__ projW,
                const float* __restrict__ projB,
                const void*  __restrict__ ws,
                float* __restrict__ out, int B) {
  __shared__ float sW[108];
  __shared__ float sPB[6];
  __shared__ float sTrig[36];
  // Per-wave 16 KB region: first used as f16 A-staging (32 rows x 130 halves),
  // then reused as f32 transformed-state buffer (32 rows x 64 complex).
  __shared__ float sBuf[2][32*128];

  const int tid  = threadIdx.x;
  const int lane = tid & 31;
  const int wave = tid >> 5;

  for (int i = tid; i < 108; i += 64) sW[i] = projW[i];
  if (tid < 6)  sPB[tid]  = projB[tid];
  const float* wsTrig = (const float*)((const char*)ws + 32768);
  if (tid < 36) sTrig[tid] = wsTrig[tid];
  __syncthreads();

  const int row  = blockIdx.x*64 + wave*32 + lane;
  const int rowc = (row < B) ? row : (B - 1);

  // ---- angles + per-qubit 2-vectors (product state factors) ----
  float x[18];
  __builtin_prefetch(&qae[rowc*18], 0, 1);
  #pragma unroll
  for (int j = 0; j < 18; ++j) x[j] = qae[rowc*18 + j];

  float u0r[NQ], u0i[NQ], u1r[NQ], u1i[NQ];
  #pragma unroll
  for (int q = 0; q < NQ; ++q) {
    float t = sPB[q];
    #pragma unroll
    for (int j = 0; j < 18; ++j) t += sW[q*18 + j] * x[j];
    float h  = tanhf(t) * (PI_F * 0.5f);
    float r0, r1;
    __sincosf(h, &r1, &r0);              // RY(a)|0> = (cos h, sin h)
    float cx = sTrig[q*6+0], sx = sTrig[q*6+1];
    float cy = sTrig[q*6+2], sy = sTrig[q*6+3];
    float cz = sTrig[q*6+4], sz = sTrig[q*6+5];
    // RX(w0)
    float a0r = cx*r0, a0i = -sx*r1;
    float a1r = cx*r1, a1i = -sx*r0;
    // RY(w1)
    float b0r = cy*a0r - sy*a1r, b0i = cy*a0i - sy*a1i;
    float b1r = sy*a0r + cy*a1r, b1i = sy*a0i + cy*a1i;
    // RZ(w2): |0> *= (cz - i sz), |1> *= (cz + i sz)
    u0r[q] = b0r*cz + b0i*sz;  u0i[q] = b0i*cz - b0r*sz;
    u1r[q] = b1r*cz - b1i*sz;  u1i[q] = b1i*cz + b1r*sz;
  }

  // ---- 8x8 split of the 64-amplitude product state ----
  float hr[8], hii[8], lr[8], li[8];
  #pragma unroll
  for (int t = 0; t < 8; ++t) {
    int b0 = (t>>2)&1, b1 = (t>>1)&1, b2 = t&1;
    float pr = b0 ? u1r[0] : u0r[0], pi = b0 ? u1i[0] : u0i[0];
    float qr = b1 ? u1r[1] : u0r[1], qi = b1 ? u1i[1] : u0i[1];
    float tr = pr*qr - pi*qi, ti = pr*qi + pi*qr;
    qr = b2 ? u1r[2] : u0r[2]; qi = b2 ? u1i[2] : u0i[2];
    hr[t]  = tr*qr - ti*qi;  hii[t] = tr*qi + ti*qr;

    float pr2 = b0 ? u1r[3] : u0r[3], pi2 = b0 ? u1i[3] : u0i[3];
    float qr2 = b1 ? u1r[4] : u0r[4], qi2 = b1 ? u1i[4] : u0i[4];
    float tr2 = pr2*qr2 - pi2*qi2, ti2 = pr2*qi2 + pi2*qr2;
    qr2 = b2 ? u1r[5] : u0r[5]; qi2 = b2 ? u1i[5] : u0i[5];
    lr[t] = tr2*qr2 - ti2*qi2;  li[t] = tr2*qi2 + ti2*qr2;
  }

  // ---- stage psi into LDS as f16 in WMMA-A K-layout: k=i -> Re, k=64+i -> Im ----
  _Float16* sA = (_Float16*)sBuf[wave];
  const int rbase = lane * 130;  // 130-half row stride: conflict-free lane stores
  #pragma unroll
  for (int i = 0; i < 64; i += 2) {
    int h0 = i>>3, l0 = i&7, h1 = (i+1)>>3, l1 = (i+1)&7;
    float p0r = hr[h0]*lr[l0] - hii[h0]*li[l0];
    float p0i = hr[h0]*li[l0] + hii[h0]*lr[l0];
    float p1r = hr[h1]*lr[l1] - hii[h1]*li[l1];
    float p1i = hr[h1]*li[l1] + hii[h1]*lr[l1];
    v2h re; re[0] = (_Float16)p0r; re[1] = (_Float16)p1r;
    v2h im; im[0] = (_Float16)p0i; im[1] = (_Float16)p1i;
    *(v2h*)&sA[rbase + i]      = re;
    *(v2h*)&sA[rbase + 64 + i] = im;
  }
  __syncthreads();

  // ---- load all 8 A fragments (2 row-tiles x 4 K-steps) ----
  const int hi8 = (lane >> 4) & 1;
  v16h afr[2][4];
  #pragma unroll
  for (int t = 0; t < 2; ++t) {
    int m = t*16 + (lane & 15);
    #pragma unroll
    for (int ks = 0; ks < 4; ++ks) {
      v16h f;
      #pragma unroll
      for (int v = 0; v < 8; ++v) {
        int kb = ks*32 + ((v & 4) ? 16 : 0) + hi8*8 + ((v & 3) << 1);
        v2h p = *(const v2h*)&sA[m*130 + kb];
        f[2*v] = p[0]; f[2*v+1] = p[1];
      }
      afr[t][ks] = f;
    }
  }
  __syncthreads();  // A region is about to be reused as f32 output buffer

  // ---- complex GEMM via WMMA: 2 comps x 4 N-tiles x 4 K-steps x 2 row-tiles ----
  const _Float16* wsB = (const _Float16*)ws;
  float* oBuf = sBuf[wave];
  #pragma unroll
  for (int comp = 0; comp < 2; ++comp) {
    #pragma unroll
    for (int nt = 0; nt < 4; ++nt) {
      v8f acc0 = {0.f,0.f,0.f,0.f,0.f,0.f,0.f,0.f};
      v8f acc1 = {0.f,0.f,0.f,0.f,0.f,0.f,0.f,0.f};
      #pragma unroll
      for (int ks = 0; ks < 4; ++ks) {
        int fragId = (comp*4 + nt)*4 + ks;
        v16h bfr = *(const v16h*)(wsB + fragId*512 + lane*16);  // coalesced 32B/lane
        acc0 = __builtin_amdgcn_wmma_f32_16x16x32_f16(
                 false, afr[0][ks], false, bfr, (short)0, acc0, false, false);
        acc1 = __builtin_amdgcn_wmma_f32_16x16x32_f16(
                 false, afr[1][ks], false, bfr, (short)0, acc1, false, false);
      }
      int nCol = nt*16 + (lane & 15);
      #pragma unroll
      for (int r = 0; r < 8; ++r) {
        int m0 = r + hi8*8;                    // C/D layout: M = r (+8 for hi half)
        oBuf[m0*128       + 2*nCol + comp] = acc0[r];
        oBuf[(16+m0)*128  + 2*nCol + comp] = acc1[r];
      }
    }
  }
  __syncthreads();

  // ---- expectation values: each lane reduces its own row ----
  if (row < B) {
    const float* pr = oBuf + lane*128;  // interleaved (re,im) x 64
    float res[18];
    #pragma unroll
    for (int q = 0; q < NQ; ++q) {
      int mask = 1 << (5 - q);
      int low  = mask - 1;
      float xr = 0.f, xi = 0.f, zz = 0.f;
      #pragma unroll 8
      for (int p = 0; p < 32; ++p) {
        int pp = (p + lane) & 31;  // lane-rotated order -> conflict-free banks
        int n0 = ((pp & ~low) << 1) | (pp & low);
        int n1 = n0 | mask;
        float c0r = pr[2*n0], c0i = pr[2*n0 + 1];
        float c1r = pr[2*n1], c1i = pr[2*n1 + 1];
        xr += c0r*c1r + c0i*c1i;
        xi += c0r*c1i - c0i*c1r;
        zz += c0r*c0r + c0i*c0i - c1r*c1r - c1i*c1i;
      }
      res[q]      = 2.f * xr;
      res[6 + q]  = 2.f * xi;
      res[12 + q] = zz;
    }
    #pragma unroll
    for (int j = 0; j < 18; ++j) out[row*18 + j] = res[j];
  }
}

extern "C" void kernel_launch(void* const* d_in, const int* in_sizes, int n_in,
                              void* d_out, int out_size, void* d_ws, size_t ws_size,
                              hipStream_t stream) {
  const float* qae   = (const float*)d_in[0];
  const float* projW = (const float*)d_in[1];
  const float* projB = (const float*)d_in[2];
  const float* pqcW  = (const float*)d_in[3];
  float* out = (float*)d_out;
  const int B = in_sizes[0] / 18;

  // ws usage: 32768 B of f16 B-fragments + 144 B trig table.
  qqk_setup_kernel<<<1, 64, 0, stream>>>(pqcW, d_ws);
  const int blocks = (B + 63) / 64;
  qqk_main_kernel<<<blocks, 64, 0, stream>>>(qae, projW, projB, d_ws, out, B);
}
